// ContextQueryAttentionLayer2_77129022701809
// MI455X (gfx1250) — compile-verified, matrix operand order checked by reading, not run on hardware
//
#include <hip/hip_runtime.h>
#include <hip/hip_bf16.h>

// ---------------------------------------------------------------------------
// Context-Query attention (BiDAF trilinear) for MI455X / gfx1250.
//   - all GEMMs on V_WMMA_F32_16X16X32_BF16 (f32 accumulate, f32 softmax)
//   - q2c computed as S_ @ (S_T @ c): 6x fewer FLOPs than reference order
//   - S (32MB) and T (4MB, bf16) live in the 192MB L2; HBM traffic is
//     dominated by c (67MB) and out (256MB) -> ~14us bandwidth floor.
//   - staging uses b128 global loads, double-buffered LDS slabs, and
//     global_prefetch_b8 via __builtin_prefetch.
// ---------------------------------------------------------------------------

typedef __bf16 bf16;
typedef __attribute__((ext_vector_type(16))) __bf16 v16bf;
typedef __attribute__((ext_vector_type(8)))  float  v8f;

constexpr int B_ = 64;
constexpr int N_ = 1024;
constexpr int M_ = 128;
constexpr int D_ = 256;
constexpr float MASKV = -1e30f;

constexpr int LPAD = 40;   // LDS row stride (elements) for 32-wide K slabs
constexpr int ASTR = 136;  // LDS row stride for the 128-wide softmax tile

__device__ __forceinline__ v8f wmma_bf16(v16bf a, v16bf b, v8f c) {
  return __builtin_amdgcn_wmma_f32_16x16x32_bf16(false, a, false, b, (short)0, c,
                                                 false, false);
}

// 16x32 bf16 operand fragment from LDS; `base` points at [lane&15][k0].
// ISA K-pair layout: VGPR r, half h = lane>>4 : k = (r<4) ? 2r+8h : 16+2(r-4)+8h
// (compiler merges these into two ds_load_b128 per fragment)
__device__ __forceinline__ v16bf load_frag(const bf16* base) {
  const int h = (threadIdx.x & 31) >> 4;
  union { v16bf v; unsigned u[8]; } f;
#pragma unroll
  for (int r = 0; r < 8; ++r) {
    const int kk = (r < 4) ? (2 * r + 8 * h) : (8 + 2 * r + 8 * h);
    f.u[r] = *(const unsigned*)(base + kk);
  }
  return f.v;
}

union bf4 { uint2 u; bf16 h[4]; };

__device__ __forceinline__ void store4(bf16* dst, float a, float b, float c, float d) {
  bf4 p;
  p.h[0] = (bf16)a; p.h[1] = (bf16)b; p.h[2] = (bf16)c; p.h[3] = (bf16)d;
  *(uint2*)dst = p.u;
}

// ---------------------------------------------------------------------------
// Kernel 1: S[b,n,m] = (c*wm)·q + c·w0 + q·w1 + bias[m]
// grid (N/128, B), block 256; double-buffered K slabs of 32.
// ---------------------------------------------------------------------------
__global__ __launch_bounds__(256)
void k_similarity(const float* __restrict__ c, const float* __restrict__ q,
                  const float* __restrict__ w0, const float* __restrict__ w1,
                  const float* __restrict__ wm, const float* __restrict__ bias,
                  float* __restrict__ S) {
  __shared__ bf16 cs[2][128 * LPAD];
  __shared__ bf16 qs[2][128 * LPAD];
  __shared__ float r0[128];
  __shared__ float q1[128];

  const int b = blockIdx.y;
  const int n0 = blockIdx.x * 128;
  const int t = threadIdx.x;
  const int wave = t >> 5, lane = t & 31, row16 = lane & 15, half = lane >> 4;

  // rank-1 terms: r0[n] = c[n]·w0 ; q1[m] = q[m]·w1 + bias[m]  (float4 dots)
  if (t < 128) {
    const float4* cr = (const float4*)(c + ((size_t)(b * N_ + n0 + t)) * D_);
    const float4* wv = (const float4*)w0;
    float acc = 0.f;
    for (int d = 0; d < D_ / 4; ++d) {
      float4 a = cr[d], w = wv[d];
      acc += a.x * w.x + a.y * w.y + a.z * w.z + a.w * w.w;
    }
    r0[t] = acc;
  } else {
    const int m = t - 128;
    const float4* qr = (const float4*)(q + ((size_t)(b * M_ + m)) * D_);
    const float4* wv = (const float4*)w1;
    float acc = 0.f;
    for (int d = 0; d < D_ / 4; ++d) {
      float4 a = qr[d], w = wv[d];
      acc += a.x * w.x + a.y * w.y + a.z * w.z + a.w * w.w;
    }
    q1[m] = acc + bias[m];
  }

  auto stage = [&](int k0, int buf) {
    for (int i = t; i < 128 * 8; i += 256) {       // 1024 float4 per slab
      const int r = i >> 3, k = (i & 7) * 4;
      float4 cv = *(const float4*)&c[((size_t)(b * N_ + n0 + r)) * D_ + k0 + k];
      float4 wv = *(const float4*)&wm[k0 + k];
      store4(&cs[buf][r * LPAD + k], cv.x * wv.x, cv.y * wv.y, cv.z * wv.z, cv.w * wv.w);
      float4 qv = *(const float4*)&q[((size_t)(b * M_ + r)) * D_ + k0 + k];
      store4(&qs[buf][r * LPAD + k], qv.x, qv.y, qv.z, qv.w);
    }
  };

  v8f acc[8];
#pragma unroll
  for (int j = 0; j < 8; ++j) acc[j] = (v8f)(0.f);

  stage(0, 0);
  __syncthreads();
  for (int kb = 0; kb < 8; ++kb) {
    if (kb + 1 < 8) stage((kb + 1) * 32, (kb + 1) & 1);   // overlap with WMMA
    const bf16* cb = cs[kb & 1];
    const bf16* qb = qs[kb & 1];
    const v16bf fa = load_frag(&cb[(wave * 16 + row16) * LPAD]);
#pragma unroll
    for (int j = 0; j < 8; ++j) {
      const v16bf fb = load_frag(&qb[(j * 16 + row16) * LPAD]);
      acc[j] = wmma_bf16(fa, fb, acc[j]);
    }
    __syncthreads();
  }

#pragma unroll
  for (int j = 0; j < 8; ++j)
#pragma unroll
    for (int r = 0; r < 8; ++r) {
      const int nl = wave * 16 + r + 8 * half;
      const int m = j * 16 + row16;
      S[((size_t)(b * N_ + n0 + nl)) * M_ + m] = acc[j][r] + r0[nl] + q1[m];
    }
}

// ---------------------------------------------------------------------------
// Kernel 2a/2b: masked column softmax stats over n, two-level reduction.
// ---------------------------------------------------------------------------
__global__ __launch_bounds__(128)
void k_colpart(const float* __restrict__ S, const int* __restrict__ c_mask,
               float* __restrict__ pmax, float* __restrict__ psum) {
  const int b = blockIdx.y, tile = blockIdx.x, m = threadIdx.x;
  float mx = -3.0e38f, sm = 0.f;
  for (int nn = 0; nn < 128; ++nn) {
    const int n = tile * 128 + nn;
    __builtin_prefetch(&S[((size_t)(b * N_ + n + 8)) * M_ + m], 0, 0);
    float v = S[((size_t)(b * N_ + n)) * M_ + m];
    if (!c_mask[b * N_ + n]) v += MASKV;
    if (v > mx) { sm *= __expf(mx - v); mx = v; }
    sm += __expf(v - mx);
  }
  pmax[(b * 8 + tile) * M_ + m] = mx;
  psum[(b * 8 + tile) * M_ + m] = sm;
}

__global__ __launch_bounds__(128)
void k_colreduce(const float* __restrict__ pmax, const float* __restrict__ psum,
                 float* __restrict__ colmax, float* __restrict__ colsum) {
  const int b = blockIdx.x, m = threadIdx.x;
  float mx = -3.0e38f;
  for (int tI = 0; tI < 8; ++tI) {
    const float v = pmax[(b * 8 + tI) * M_ + m];
    mx = v > mx ? v : mx;
  }
  float sm = 0.f;
  for (int tI = 0; tI < 8; ++tI)
    sm += psum[(b * 8 + tI) * M_ + m] * __expf(pmax[(b * 8 + tI) * M_ + m] - mx);
  colmax[b * M_ + m] = mx;
  colsum[b * M_ + m] = sm;
}

// ---------------------------------------------------------------------------
// Kernel 3: T[b,m,d] = sum_n softmax_col(S)[n,m] * c[b,n,d]  -> bf16
// grid (D/128, B), block 256; double-buffered K(n) slabs of 32.
// ---------------------------------------------------------------------------
__global__ __launch_bounds__(256)
void k_t(const float* __restrict__ S, const float* __restrict__ c,
         const int* __restrict__ c_mask, const float* __restrict__ colmax,
         const float* __restrict__ colsum, bf16* __restrict__ Tb) {
  __shared__ bf16 as[2][128 * LPAD];   // A[m][k=n_local]  (col-softmax of S^T)
  __shared__ bf16 bs[2][128 * LPAD];   // B[d_local][k=n_local] (transposed c)
  __shared__ float cmx[128], rcs[128];

  const int b = blockIdx.y;
  const int d0 = blockIdx.x * 128;
  const int t = threadIdx.x;
  const int wave = t >> 5, lane = t & 31, row16 = lane & 15, half = lane >> 4;

  if (t < 128) {
    cmx[t] = colmax[b * M_ + t];
    rcs[t] = 1.f / colsum[b * M_ + t];
  }
  __syncthreads();

  auto stage = [&](int n0, int buf) {
    for (int i = t; i < 32 * 32; i += 256) {       // 32 rows x 32 float4
      const int nn = i >> 5, x = (i & 31) * 4;
      const float madd = c_mask[b * N_ + n0 + nn] ? 0.f : MASKV;
      float4 sv = *(const float4*)&S[((size_t)(b * N_ + n0 + nn)) * M_ + x];
      as[buf][(x + 0) * LPAD + nn] = (bf16)(__expf(sv.x + madd - cmx[x + 0]) * rcs[x + 0]);
      as[buf][(x + 1) * LPAD + nn] = (bf16)(__expf(sv.y + madd - cmx[x + 1]) * rcs[x + 1]);
      as[buf][(x + 2) * LPAD + nn] = (bf16)(__expf(sv.z + madd - cmx[x + 2]) * rcs[x + 2]);
      as[buf][(x + 3) * LPAD + nn] = (bf16)(__expf(sv.w + madd - cmx[x + 3]) * rcs[x + 3]);
      float4 cv = *(const float4*)&c[((size_t)(b * N_ + n0 + nn)) * D_ + d0 + x];
      bs[buf][(x + 0) * LPAD + nn] = (bf16)cv.x;
      bs[buf][(x + 1) * LPAD + nn] = (bf16)cv.y;
      bs[buf][(x + 2) * LPAD + nn] = (bf16)cv.z;
      bs[buf][(x + 3) * LPAD + nn] = (bf16)cv.w;
    }
  };

  v8f acc[8];
#pragma unroll
  for (int j = 0; j < 8; ++j) acc[j] = (v8f)(0.f);

  stage(0, 0);
  __syncthreads();
  for (int kb = 0; kb < N_ / 32; ++kb) {
    if (kb + 1 < N_ / 32) stage((kb + 1) * 32, (kb + 1) & 1);
    const bf16* ab = as[kb & 1];
    const bf16* bb = bs[kb & 1];
    const v16bf fa = load_frag(&ab[(wave * 16 + row16) * LPAD]);
#pragma unroll
    for (int j = 0; j < 8; ++j) {
      const v16bf fb = load_frag(&bb[(j * 16 + row16) * LPAD]);
      acc[j] = wmma_bf16(fa, fb, acc[j]);
    }
    __syncthreads();
  }

#pragma unroll
  for (int j = 0; j < 8; ++j)
#pragma unroll
    for (int r = 0; r < 8; ++r) {
      const int m = wave * 16 + r + 8 * half;
      const int d = d0 + j * 16 + row16;
      Tb[((size_t)(b * M_ + m)) * D_ + d] = (bf16)acc[j][r];
    }
}

// ---------------------------------------------------------------------------
// Kernel 4: row softmax -> A (128x128 bf16), c2q = A@q, q2c = A@T (shared A),
// write [c, c2q, c*c2q, c*q2c].  grid (N/128, D/128, B), block 256.
// ---------------------------------------------------------------------------
__global__ __launch_bounds__(256)
void k_final(const float* __restrict__ S, const float* __restrict__ q,
             const bf16* __restrict__ Tb, const float* __restrict__ c,
             const int* __restrict__ q_mask, float* __restrict__ out) {
  __shared__ bf16 as[128 * ASTR];   // A[n_local][m]
  __shared__ bf16 qs[128 * LPAD];   // q^T slab [d_local][k=m_local]
  __shared__ bf16 ts[128 * LPAD];   // T^T slab [d_local][k=m_local]
  __shared__ float pm[256], ps[256];

  const int b = blockIdx.z;
  const int n0 = blockIdx.x * 128;
  const int d0 = blockIdx.y * 128;
  const int t = threadIdx.x;
  const int wave = t >> 5, lane = t & 31, row16 = lane & 15, half = lane >> 4;

  // masked row softmax, 2 threads per row (each scans 64 m values)
  {
    const int row = t & 127, hh = t >> 7;
    const float* sr = S + ((size_t)(b * N_ + n0 + row)) * M_;
    const int mlo = hh * 64;
    float mx = -3.0e38f;
    for (int m = mlo; m < mlo + 64; ++m) {
      float v = sr[m];
      if (!q_mask[b * M_ + m]) v += MASKV;
      mx = v > mx ? v : mx;
    }
    pm[t] = mx;
    __syncthreads();
    const float rmx = pm[row] > pm[row + 128] ? pm[row] : pm[row + 128];
    float sm = 0.f;
    for (int m = mlo; m < mlo + 64; ++m) {
      float v = sr[m];
      if (!q_mask[b * M_ + m]) v += MASKV;
      sm += __expf(v - rmx);
    }
    ps[t] = sm;
    __syncthreads();
    const float inv = 1.f / (ps[row] + ps[row + 128]);
    for (int m = mlo; m < mlo + 64; ++m) {
      float v = sr[m];
      if (!q_mask[b * M_ + m]) v += MASKV;
      as[row * ASTR + m] = (bf16)(__expf(v - rmx) * inv);
    }
  }

  v8f acc1[8], acc2[8];
#pragma unroll
  for (int j = 0; j < 8; ++j) { acc1[j] = (v8f)(0.f); acc2[j] = (v8f)(0.f); }

  __syncthreads();
  for (int m0 = 0; m0 < M_; m0 += 32) {
    for (int i = t; i < 32 * 32; i += 256) {       // 32 m-rows x 32 chunks of 4 d
      const int mm = i >> 5, d = (i & 31) * 4;
      if (m0 + 32 < M_ && (i & 31) == 0) {          // gfx1250 global_prefetch_b8
        __builtin_prefetch(&q[((size_t)(b * M_ + m0 + 32 + mm)) * D_ + d0], 0, 0);
        __builtin_prefetch(&Tb[((size_t)(b * M_ + m0 + 32 + mm)) * D_ + d0], 0, 0);
      }
      float4 qv = *(const float4*)&q[((size_t)(b * M_ + m0 + mm)) * D_ + d0 + d];
      qs[(d + 0) * LPAD + mm] = (bf16)qv.x;
      qs[(d + 1) * LPAD + mm] = (bf16)qv.y;
      qs[(d + 2) * LPAD + mm] = (bf16)qv.z;
      qs[(d + 3) * LPAD + mm] = (bf16)qv.w;
      bf4 tv;
      tv.u = *(const uint2*)&Tb[((size_t)(b * M_ + m0 + mm)) * D_ + d0 + d];
      ts[(d + 0) * LPAD + mm] = tv.h[0];
      ts[(d + 1) * LPAD + mm] = tv.h[1];
      ts[(d + 2) * LPAD + mm] = tv.h[2];
      ts[(d + 3) * LPAD + mm] = tv.h[3];
    }
    __syncthreads();

    const v16bf fa = load_frag(&as[(wave * 16 + row16) * ASTR + m0]);
#pragma unroll
    for (int j = 0; j < 8; ++j) {
      const v16bf fbq = load_frag(&qs[(j * 16 + row16) * LPAD]);
      acc1[j] = wmma_bf16(fa, fbq, acc1[j]);
      const v16bf fbt = load_frag(&ts[(j * 16 + row16) * LPAD]);
      acc2[j] = wmma_bf16(fa, fbt, acc2[j]);
    }
    __syncthreads();
  }

#pragma unroll
  for (int j = 0; j < 8; ++j)
#pragma unroll
    for (int r = 0; r < 8; ++r) {
      const int n = n0 + wave * 16 + r + 8 * half;
      const int d = d0 + j * 16 + row16;
      const float cv = c[((size_t)(b * N_ + n)) * D_ + d];
      const float c2q = acc1[j][r];
      const float q2c = acc2[j][r];
      float* o = out + ((size_t)(b * N_ + n)) * (4 * D_);
      o[d]          = cv;
      o[D_ + d]     = c2q;
      o[2 * D_ + d] = cv * c2q;
      o[3 * D_ + d] = cv * q2c;
    }
}

// ---------------------------------------------------------------------------
extern "C" void kernel_launch(void* const* d_in, const int* in_sizes, int n_in,
                              void* d_out, int out_size, void* d_ws, size_t ws_size,
                              hipStream_t stream) {
  const float* c      = (const float*)d_in[0];
  const float* q      = (const float*)d_in[1];
  const int*   c_mask = (const int*)d_in[2];
  const int*   q_mask = (const int*)d_in[3];
  const float* w0     = (const float*)d_in[4];
  const float* w1     = (const float*)d_in[5];
  const float* wm     = (const float*)d_in[6];
  const float* bias   = (const float*)d_in[7];
  float* out = (float*)d_out;

  float* S      = (float*)d_ws;                          // B*N*M f32 (32 MB)
  float* pmax   = S + (size_t)B_ * N_ * M_;              // B*8*M
  float* psum   = pmax + (size_t)B_ * 8 * M_;            // B*8*M
  float* colmax = psum + (size_t)B_ * 8 * M_;            // B*M
  float* colsum = colmax + (size_t)B_ * M_;              // B*M
  bf16*  Tb     = (bf16*)(colsum + (size_t)B_ * M_);     // B*M*D bf16 (4 MB)

  k_similarity<<<dim3(N_ / 128, B_), 256, 0, stream>>>(c, q, w0, w1, wm, bias, S);
  k_colpart<<<dim3(N_ / 128, B_), 128, 0, stream>>>(S, c_mask, pmax, psum);
  k_colreduce<<<dim3(B_), 128, 0, stream>>>(pmax, psum, colmax, colsum);
  k_t<<<dim3(D_ / 128, B_), 256, 0, stream>>>(S, c, c_mask, colmax, colsum, Tb);
  k_final<<<dim3(N_ / 128, D_ / 128, B_), 256, 0, stream>>>(S, q, Tb, c, q_mask, out);
}
